// Optim4RL_19705309954329
// MI455X (gfx1250) — matrix operand order repaired and checked by reading, not compiled, not run on hardware
//
#include <hip/hip_runtime.h>
#include <hip/hip_bf16.h>

typedef _Float16 f16t;
typedef __attribute__((ext_vector_type(16))) _Float16 v16h;
typedef __attribute__((ext_vector_type(8)))  float    v8f;

#define L2E 1.44269504088896340f

static __device__ __forceinline__ v8f wmma16(v16h a, v16h b, v8f c) {
    // D = A(16x32 f16) x B(32x16 f16) + C(16x16 f32)
    return __builtin_amdgcn_wmma_f32_16x16x32_f16(false, a, false, b, (short)0, c, false, false);
}

static __device__ __forceinline__ float fexp(float x)  { return __builtin_amdgcn_exp2f(x * L2E); }
static __device__ __forceinline__ float frcp(float x)  { return __builtin_amdgcn_rcpf(x); }
static __device__ __forceinline__ float frsq(float x)  { return __builtin_amdgcn_rsqf(x); }
static __device__ __forceinline__ float fsigmoid(float x) { return frcp(1.0f + fexp(-x)); }
static __device__ __forceinline__ float ftanh(float x)    { return 1.0f - 2.0f * frcp(1.0f + fexp(2.0f * x)); }

// B layout (32x16, f16): element e of v16h (h = lane/16): e<8 -> K-row 8h+e, e>=8 -> K-row 16+8h+(e-8);
// column = lane % 16.

// GRU r|z tile: rows 0-7 = Wh (8x24) cols 0-15, row 8 = Wi (so xi folds into the WMMA), rest 0.
static __device__ __forceinline__ v16h make_b_gru_rz(const float* __restrict__ Wh,
                                                     const float* __restrict__ Wi, int lane) {
    int col = lane & 15;
    int rb  = (lane >= 16) ? 16 : 0;
    v16h b;
#pragma unroll
    for (int e = 0; e < 16; ++e) {
        int r = (e < 8) ? (rb / 2 + e) : (16 + rb / 2 + e - 8);
        float v = 0.0f;
        if (r < 8)       v = Wh[r * 24 + col];
        else if (r == 8) v = Wi[col];
        b[e] = (f16t)v;
    }
    return b;
}

// GRU n tile: cols 0-7: rows 0-7 = Wh_n (cols 16-23)  -> D = hh_n
//             cols 8-15: row 8    = Wi_n               -> D = g * Wi_n = xi_n
static __device__ __forceinline__ v16h make_b_gru_n(const float* __restrict__ Wh,
                                                    const float* __restrict__ Wi, int lane) {
    int col = lane & 15;
    int rb  = (lane >= 16) ? 16 : 0;
    v16h b;
#pragma unroll
    for (int e = 0; e < 16; ++e) {
        int r = (e < 8) ? (rb / 2 + e) : (16 + rb / 2 + e - 8);
        float v = 0.0f;
        if (r < 8 && col < 8)        v = Wh[r * 24 + 16 + col];
        else if (r == 8 && col >= 8) v = Wi[16 + (col - 8)];
        b[e] = (f16t)v;
    }
    return b;
}

// MLP weight tile: rows 0..K-1 = W (K x 16), rest 0.
static __device__ __forceinline__ v16h make_b_mlp(const float* __restrict__ W, int K, int lane) {
    int col = lane & 15;
    int rb  = (lane >= 16) ? 16 : 0;
    v16h b;
#pragma unroll
    for (int e = 0; e < 16; ++e) {
        int r = (e < 8) ? (rb / 2 + e) : (16 + rb / 2 + e - 8);
        b[e] = (f16t)((r < K) ? W[r * 16 + col] : 0.0f);
    }
    return b;
}

// Identity (8x16): moves h_old from A-layout into D-layout via one WMMA.
static __device__ __forceinline__ v16h make_b_ident(int lane) {
    int col = lane & 15;
    int rb  = (lane >= 16) ? 16 : 0;
    v16h b;
#pragma unroll
    for (int e = 0; e < 16; ++e) {
        int r = (e < 8) ? (rb / 2 + e) : (16 + rb / 2 + e - 8);
        b[e] = (f16t)((r < 8 && col == r) ? 1.0f : 0.0f);
    }
    return b;
}

// A layout (16x32, f16): lanes 0-15: elements 0-7 = K0-7 (h features of param=lane),
// lanes 16-31: element 0 = K8 (the GRU input x), everything else 0.
static __device__ __forceinline__ v16h make_a_gru(float4 a0, float4 a1, float x, int lane) {
    float av[16];
#pragma unroll
    for (int i = 0; i < 16; ++i) av[i] = 0.0f;
    if (lane < 16) {
        av[0] = a0.x; av[1] = a0.y; av[2] = a0.z; av[3] = a0.w;
        av[4] = a1.x; av[5] = a1.y; av[6] = a1.z; av[7] = a1.w;
    } else {
        av[0] = x;
    }
    v16h a;
#pragma unroll
    for (int i = 0; i < 16; ++i) a[i] = (f16t)av[i];
    return a;
}

// One GRU in D layout: per-v element: param = v + 8*(lane>=16), feature column = lane%16.
static __device__ __forceinline__ void gru_step(v16h A, v16h Brz, v16h Bn, v16h BI,
                                                float bhrz, float bhn, float* hnew) {
    v8f c0 = {};
    v8f drz = wmma16(A, Brz, c0);   // cols 0-7: xi_r+hh_r, cols 8-15: xi_z+hh_z
    v8f dn  = wmma16(A, Bn,  c0);   // cols 0-7: hh_n,      cols 8-15: xi_n
    v8f dh  = wmma16(A, BI,  c0);   // h_old transposed into D layout
#pragma unroll
    for (int v = 0; v < 8; ++v) {
        float rz  = fsigmoid(drz[v] + bhrz);      // cols 0-7: r, cols 8-15: z
        float z   = __shfl_xor(rz, 8, 32);        // z down to cols 0-7
        float xin = __shfl_xor(dn[v], 8, 32);     // xi_n down to cols 0-7
        float n   = ftanh(xin + rz * (dn[v] + bhn));
        hnew[v]   = n + z * (dh[v] - n);          // (1-z)*n + z*h_old
    }
}

// One MLP head 8->16->16->1 through LDS staging buffer sm (16 params x 16 feats, f32).
static __device__ __forceinline__ void mlp_head(float* sm, v16h B1, v16h B2,
                                                float b1v, float b2v, float w3v, float b3,
                                                int lane, float* o) {
    int pl  = lane & 15;
    int hi8 = (lane >= 16) ? 8 : 0;
    v8f c0 = {};

    // Layer 1: A from h_new (8 features)
    float av[16];
#pragma unroll
    for (int i = 0; i < 16; ++i) av[i] = 0.0f;
    if (lane < 16) {
        float4 y0 = *(const float4*)(sm + pl * 16);
        float4 y1 = *(const float4*)(sm + pl * 16 + 4);
        av[0] = y0.x; av[1] = y0.y; av[2] = y0.z; av[3] = y0.w;
        av[4] = y1.x; av[5] = y1.y; av[6] = y1.z; av[7] = y1.w;
    }
    v16h a1;
#pragma unroll
    for (int i = 0; i < 16; ++i) a1[i] = (f16t)av[i];
    v8f d1 = wmma16(a1, B1, c0);

    // relu + restage 16-wide activations
    __builtin_amdgcn_wave_barrier();
#pragma unroll
    for (int v = 0; v < 8; ++v) {
        float x = fmaxf(d1[v] + b1v, 0.0f);
        sm[(v + hi8) * 16 + (lane & 15)] = x;
    }
    __builtin_amdgcn_wave_barrier();

    // Layer 2: A from 16 features (lanes<16: feats 0-7 = K0-7, lanes>=16: feats 8-15 = K8-15)
    int fb = (lane >= 16) ? 8 : 0;
    float4 z0 = *(const float4*)(sm + pl * 16 + fb);
    float4 z1 = *(const float4*)(sm + pl * 16 + fb + 4);
    float av2[16];
#pragma unroll
    for (int i = 0; i < 16; ++i) av2[i] = 0.0f;
    av2[0] = z0.x; av2[1] = z0.y; av2[2] = z0.z; av2[3] = z0.w;
    av2[4] = z1.x; av2[5] = z1.y; av2[6] = z1.z; av2[7] = z1.w;
    v16h a2;
#pragma unroll
    for (int i = 0; i < 16; ++i) a2[i] = (f16t)av2[i];
    v8f d2 = wmma16(a2, B2, c0);

    // Layer 3 (16 -> 1): relu, scale by W3[col], butterfly-reduce across the 16-lane half.
#pragma unroll
    for (int v = 0; v < 8; ++v) {
        float t = fmaxf(d2[v] + b2v, 0.0f) * w3v;
        t += __shfl_xor(t, 1, 32);
        t += __shfl_xor(t, 2, 32);
        t += __shfl_xor(t, 4, 32);
        t += __shfl_xor(t, 8, 32);
        o[v] = t + b3;
    }
}

__global__ void __launch_bounds__(256)
optim4rl_kernel(const float* __restrict__ h, const float* __restrict__ g,
                const float* __restrict__ Wi1, const float* __restrict__ Wh1, const float* __restrict__ bh1,
                const float* __restrict__ Wi2, const float* __restrict__ Wh2, const float* __restrict__ bh2,
                const float* __restrict__ m1W0, const float* __restrict__ m1W1, const float* __restrict__ m1W2,
                const float* __restrict__ m1b0, const float* __restrict__ m1b1, const float* __restrict__ m1b2,
                const float* __restrict__ m2W0, const float* __restrict__ m2W1, const float* __restrict__ m2W2,
                const float* __restrict__ m2b0, const float* __restrict__ m2b1, const float* __restrict__ m2b2,
                float* __restrict__ outH, float* __restrict__ outO, long N)
{
    __shared__ float smem[8][2][256];   // per wave: two 16x16 f32 staging tiles

    const int lane = threadIdx.x & 31;
    const int wid  = threadIdx.x >> 5;
    const int c    = lane & 15;
    const int hi   = (lane >= 16) ? 1 : 0;
    const int cm8  = (c < 8) ? c : 7;

    // ---- weight tiles (built once per wave, live in VGPRs) ----
    v16h Bg1rz = make_b_gru_rz(Wh1, Wi1, lane);
    v16h Bg1n  = make_b_gru_n (Wh1, Wi1, lane);
    v16h Bg2rz = make_b_gru_rz(Wh2, Wi2, lane);
    v16h Bg2n  = make_b_gru_n (Wh2, Wi2, lane);
    v16h BI    = make_b_ident(lane);
    v16h Bm10  = make_b_mlp(m1W0, 8,  lane);
    v16h Bm11  = make_b_mlp(m1W1, 16, lane);
    v16h Bm20  = make_b_mlp(m2W0, 8,  lane);
    v16h Bm21  = make_b_mlp(m2W1, 16, lane);

    // ---- per-lane broadcast vectors ----
    const float bh1rz = bh1[c], bh1nv = bh1[16 + cm8];
    const float bh2rz = bh2[c], bh2nv = bh2[16 + cm8];
    const float b10 = m1b0[c], b11v = m1b1[c], w31 = m1W2[c], b31 = m1b2[0];
    const float b20 = m2b0[c], b21v = m2b1[c], w32 = m2W2[c], b32 = m2b2[0];

    const float* h1p = h;
    const float* h2p = h + (size_t)N * 8;
    float* hO1 = outH;
    float* hO2 = outH + (size_t)N * 8;

    float* sm0 = &smem[wid][0][0];
    float* sm1 = &smem[wid][1][0];

    const long ntiles = N >> 4;
    const int  gwave  = blockIdx.x * (blockDim.x >> 5) + wid;
    const int  nwaves = gridDim.x * (blockDim.x >> 5);

    for (long t = gwave; t < ntiles; t += nwaves) {
        const long base = t << 4;
        const long p    = base + c;

        const float gl   = g[p];
        const float gsel = g[base + (c & 7) + 8 * hi];   // g for param (c&7)+8*hi (sign at gather lanes)
        float4 hA0 = {0,0,0,0}, hA1 = {0,0,0,0}, hB0 = {0,0,0,0}, hB1 = {0,0,0,0};
        if (lane < 16) {
            const float4* q1 = (const float4*)(h1p + p * 8);
            hA0 = q1[0]; hA1 = q1[1];
            const float4* q2 = (const float4*)(h2p + p * 8);
            hB0 = q2[0]; hB1 = q2[1];
        }
        v16h A1 = make_a_gru(hA0, hA1, gl, lane);
        v16h A2 = make_a_gru(hB0, hB1, gl * gl, lane);

        float h1n[8], h2n[8];
        gru_step(A1, Bg1rz, Bg1n, BI, bh1rz, bh1nv, h1n);
        gru_step(A2, Bg2rz, Bg2n, BI, bh2rz, bh2nv, h2n);

        // stage h_new into LDS (feats 0-7 valid), then write coalesced b128 to global
        __builtin_amdgcn_wave_barrier();
#pragma unroll
        for (int v = 0; v < 8; ++v) {
            const int prm = v + hi * 8;
            if (c < 8) {
                sm0[prm * 16 + c] = h1n[v];
                sm1[prm * 16 + c] = h2n[v];
            }
        }
        __builtin_amdgcn_wave_barrier();
        if (lane < 16) {
            float4 s0 = *(const float4*)(sm0 + lane * 16);
            float4 s1 = *(const float4*)(sm0 + lane * 16 + 4);
            float4 s2 = *(const float4*)(sm1 + lane * 16);
            float4 s3 = *(const float4*)(sm1 + lane * 16 + 4);
            float4* q1 = (float4*)(hO1 + (base + lane) * 8);
            q1[0] = s0; q1[1] = s1;
            float4* q2 = (float4*)(hO2 + (base + lane) * 8);
            q2[0] = s2; q2[1] = s3;
        }
        __builtin_amdgcn_wave_barrier();

        float o1[8], o2[8];
        mlp_head(sm0, Bm10, Bm11, b10, b11v, w31, b31, lane, o1);
        mlp_head(sm1, Bm20, Bm21, b20, b21v, w32, b32, lane, o2);

        // out = -sign(g) * exp(o1) * rsqrt(exp(o2) + eps); gather through LDS for one b32x16 store
        __builtin_amdgcn_wave_barrier();
        const float sg = (gsel > 0.0f) ? 1.0f : ((gsel < 0.0f) ? -1.0f : 0.0f);
#pragma unroll
        for (int v = 0; v < 8; ++v) {
            const float val = -sg * fexp(o1[v]) * frsq(fexp(o2[v]) + 1e-8f);
            if (c == v) sm0[v + hi * 8] = val;   // only lanes c==v keep it; their sg matches param v+8*hi
        }
        __builtin_amdgcn_wave_barrier();
        if (lane < 16) outO[base + lane] = sm0[lane];
        __builtin_amdgcn_wave_barrier();
    }
}

extern "C" void kernel_launch(void* const* d_in, const int* in_sizes, int n_in,
                              void* d_out, int out_size, void* d_ws, size_t ws_size,
                              hipStream_t stream) {
    (void)out_size; (void)d_ws; (void)ws_size;

    const float* h   = (const float*)d_in[0];
    const float* g   = (const float*)d_in[1];
    const float* Wi1 = (const float*)d_in[2];
    const float* Wh1 = (const float*)d_in[3];
    const float* bh1 = (const float*)d_in[4];
    const float* Wi2 = (const float*)d_in[5];
    const float* Wh2 = (const float*)d_in[6];
    const float* bh2 = (const float*)d_in[7];

    const float *m1W0, *m1W1, *m1W2, *m1b0, *m1b1, *m1b2;
    const float *m2W0, *m2W1, *m2W2, *m2b0, *m2b1, *m2b2;

    if (n_in >= 20) {
        // pytree-flattened: each list element is its own input
        m1W0 = (const float*)d_in[8];  m1W1 = (const float*)d_in[9];  m1W2 = (const float*)d_in[10];
        m1b0 = (const float*)d_in[11]; m1b1 = (const float*)d_in[12]; m1b2 = (const float*)d_in[13];
        m2W0 = (const float*)d_in[14]; m2W1 = (const float*)d_in[15]; m2W2 = (const float*)d_in[16];
        m2b0 = (const float*)d_in[17]; m2b1 = (const float*)d_in[18]; m2b2 = (const float*)d_in[19];
    } else {
        // each list concatenated into one flat input
        const float* w1 = (const float*)d_in[8];
        const float* b1 = (const float*)d_in[9];
        const float* w2 = (const float*)d_in[10];
        const float* b2 = (const float*)d_in[11];
        m1W0 = w1; m1W1 = w1 + 128; m1W2 = w1 + 384;
        m1b0 = b1; m1b1 = b1 + 16;  m1b2 = b1 + 32;
        m2W0 = w2; m2W1 = w2 + 128; m2W2 = w2 + 384;
        m2b0 = b2; m2b1 = b2 + 16;  m2b2 = b2 + 32;
    }

    const long N = (long)in_sizes[1];
    float* outH = (float*)d_out;                       // (2, N, 8)
    float* outO = (float*)d_out + (size_t)2 * N * 8;   // (N,)

    long ntiles = N >> 4;
    long blocks = (ntiles + 7) / 8;     // 8 waves/block, 1 tile/wave minimum
    if (blocks > 4096) blocks = 4096;   // persistent grid-stride loop
    if (blocks < 1) blocks = 1;

    optim4rl_kernel<<<dim3((unsigned)blocks), dim3(256), 0, stream>>>(
        h, g, Wi1, Wh1, bh1, Wi2, Wh2, bh2,
        m1W0, m1W1, m1W2, m1b0, m1b1, m1b2,
        m2W0, m2W1, m2W2, m2b0, m2b1, m2b2,
        outH, outO, N);
}
